// MultiHeadLatentAttention_22351009808765
// MI455X (gfx1250) — compile-verified, matrix operand order checked by reading, not run on hardware
//
#include <hip/hip_runtime.h>
#include <hip/hip_bf16.h>
#include <stdint.h>

// ---------------------------------------------------------------------------
// MLA forward for MI455X (gfx1250).
// Strategy: one-time f32->bf16 conversion passes (bandwidth-trivial at 23.3TB/s),
// then all GEMM-shaped work runs v_wmma_f32_16x16x32_bf16 with tiles streamed
// into LDS via global_load_async_to_lds_b128 (ASYNCcnt), double-buffered.
// ---------------------------------------------------------------------------

#define SEQ       4096
#define D_IN      2048
#define NUM_HEADS 16
#define QK_NOPE   64
#define QK_ROPE   64
#define QK_HEAD   128
#define V_HEAD    128
#define KV_LORA   1024
#define KVB_N     (NUM_HEADS * (QK_NOPE + V_HEAD))   // 3072
#define KVA_N     (KV_LORA + QK_ROPE)                // 1088
#define ATT_SCALE 0.08838834764831845f               // 128^-0.5

typedef __attribute__((ext_vector_type(16))) __bf16 v16bf;
typedef __attribute__((ext_vector_type(8)))  float  v8f;
typedef __attribute__((ext_vector_type(4)))  int    v4i;

#define AS_GLOBAL __attribute__((address_space(1)))
#define AS_LDS    __attribute__((address_space(3)))

// ---- f32 -> bf16 (RNE) ------------------------------------------------------
__device__ __forceinline__ uint32_t f2bf(float a) {
    uint32_t u = __float_as_uint(a);
    return (u + 0x7FFFu + ((u >> 16) & 1u)) >> 16;
}
__device__ __forceinline__ uint32_t pack_bf16x2(float a, float b) {
    return f2bf(a) | (f2bf(b) << 16);
}

// ---- gfx1250 async global->LDS copy (16B per lane), ASYNCcnt-tracked -------
__device__ __forceinline__ void async_copy16(const void* gsrc, void* ldst) {
#if defined(__AMDGCN__) && __has_builtin(__builtin_amdgcn_global_load_async_to_lds_b128)
    __builtin_amdgcn_global_load_async_to_lds_b128(
        (AS_GLOBAL v4i*)gsrc, (AS_LDS v4i*)ldst, 0, 0);
#else
    *(uint4*)ldst = *(const uint4*)gsrc;   // fallback: through VGPRs
#endif
}
__device__ __forceinline__ void wait_async0() {
#if defined(__AMDGCN__)
#if __has_builtin(__builtin_amdgcn_s_wait_asynccnt)
    __builtin_amdgcn_s_wait_asynccnt(0);
#else
    asm volatile("s_wait_asynccnt 0" ::: "memory");
#endif
#endif
}

// ---- fragment loader for v_wmma_f32_16x16x32_bf16 ---------------------------
// 16-bit A 16x32 layout (ISA 7.12.2): lanes 0-15 row m=lane; VGPR j<4 ->
// K={2j,2j+1} (+8 for lanes 16-31); VGPR j>=4 -> K=16+...  B uses the same
// layout on rows of B^T. row_base -> row 0 of the 16-row group at the chunk.
__device__ __forceinline__ v16bf load_frag_lds(const uint32_t* row_base, int strideU32) {
    const int lane = threadIdx.x & 31;
    const int half = lane >> 4;
    const uint32_t* p = row_base + (lane & 15) * strideU32;
    union { uint32_t u[8]; v16bf v; } r;
#pragma unroll
    for (int j = 0; j < 4; ++j) r.u[j]     = p[4 * half + j];
#pragma unroll
    for (int j = 0; j < 4; ++j) r.u[4 + j] = p[8 + 4 * half + j];
    return r.v;
}

__device__ __forceinline__ v8f wmma_bf16(v16bf a, v16bf b, v8f c) {
    return __builtin_amdgcn_wmma_f32_16x16x32_bf16(
        false, a, false, b, (short)0, c, false, false);
}

// ---------------------------------------------------------------------------
// f32 -> packed bf16 pairs (one-time conversion passes)
// ---------------------------------------------------------------------------
__global__ __launch_bounds__(256)
void cvt_f32_bf16(const float* __restrict__ src, uint32_t* __restrict__ dst, int npairs) {
    int i = blockIdx.x * 256 + threadIdx.x;
    if (i < npairs) dst[i] = pack_bf16x2(src[2 * i], src[2 * i + 1]);
}

// ---------------------------------------------------------------------------
// GEMM: C[M,N] = A[M,K] @ W[N,K]^T.  A,W bf16 (u16), C f32.
// 128 threads = 4 waves; tile 64x64, K-step 32; double-buffered async loads.
// ---------------------------------------------------------------------------
__global__ __launch_bounds__(128)
void gemm_bf16_wmma(const uint16_t* __restrict__ A, const uint16_t* __restrict__ W,
                    float* __restrict__ C, int M, int N, int K) {
    __shared__ uint32_t As[2][64 * 16];   // 64 rows x 32 bf16 = 4KB per buffer
    __shared__ uint32_t Ws[2][64 * 16];

    const int t    = threadIdx.x;
    const int w    = t >> 5;
    const int lane = t & 31;
    const int half = lane >> 4;
    const int n    = lane & 15;
    const int m0   = blockIdx.y * 64;
    const int n0   = blockIdx.x * 64;

    // issue a 64x32 bf16 tile as 256 async 16B chunks (2 per thread)
    auto load_tile = [&](const uint16_t* g, int stride, int row0, int k0, uint32_t* tile) {
#pragma unroll
        for (int i = 0; i < 2; ++i) {
            int c = t + i * 128;
            int row = c >> 2, sub = c & 3;
            const uint16_t* src = g + (size_t)(row0 + row) * stride + k0 + sub * 8;
            async_copy16(src, (char*)tile + row * 64 + sub * 16);
        }
    };

    v8f acc[4];
#pragma unroll
    for (int i = 0; i < 4; ++i)
        acc[i] = (v8f){0.f, 0.f, 0.f, 0.f, 0.f, 0.f, 0.f, 0.f};

    int buf = 0;
    load_tile(A, K, m0, 0, As[0]);
    load_tile(W, K, n0, 0, Ws[0]);
    wait_async0();
    __syncthreads();

    for (int k0 = 0; k0 < K; k0 += 32) {
        if (k0 + 32 < K) {                       // prefetch next tiles (async DMA)
            load_tile(A, K, m0, k0 + 32, As[buf ^ 1]);
            load_tile(W, K, n0, k0 + 32, Ws[buf ^ 1]);
        }
        v16bf af = load_frag_lds(As[buf] + (w * 16) * 16, 16);
#pragma unroll
        for (int nt = 0; nt < 4; ++nt) {
            v16bf bf = load_frag_lds(Ws[buf] + (nt * 16) * 16, 16);
            acc[nt] = wmma_bf16(af, bf, acc[nt]);
        }
        wait_async0();       // my writes into buf^1 landed
        __syncthreads();     // everyone's done reading buf / writing buf^1
        buf ^= 1;
    }

    // D layout: lane (n, half); VGPR j -> row j + 8*half
#pragma unroll
    for (int nt = 0; nt < 4; ++nt)
#pragma unroll
        for (int j = 0; j < 8; ++j)
            C[(size_t)(m0 + w * 16 + j + 8 * half) * N + n0 + nt * 16 + n] = acc[nt][j];
}

// ---------------------------------------------------------------------------
// Fused RMSNorm(kv) -> bf16 kvnorm, RoPE(q_pe in-place, f32), RoPE(k_pe).
// ---------------------------------------------------------------------------
__global__ __launch_bounds__(256)
void rms_rope_kernel(float* __restrict__ q,              // [SEQ, 2048] f32
                     const float* __restrict__ kvfull,   // [SEQ, 1088]
                     const float* __restrict__ cosp,     // [SEQ, 32]
                     const float* __restrict__ sinp,     // [SEQ, 32]
                     const float* __restrict__ kvw,      // [1024]
                     uint32_t* __restrict__ kvnorm_bf,   // [SEQ, 512] packed bf16
                     float* __restrict__ kpe) {          // [SEQ, 64]
    __shared__ float red[256];
    const int row = blockIdx.x;
    const int t   = threadIdx.x;
    const float* kv = kvfull + (size_t)row * KVA_N;

    float ss = 0.f;
    for (int i = t; i < KV_LORA; i += 256) { float v = kv[i]; ss += v * v; }
    red[t] = ss;
    __syncthreads();
    for (int s = 128; s > 0; s >>= 1) {
        if (t < s) red[t] += red[t + s];
        __syncthreads();
    }
    const float scale = rsqrtf(red[0] * (1.f / KV_LORA) + 1e-6f);

    for (int i = t; i < KV_LORA / 2; i += 256) {
        float a = kv[2 * i]     * scale * kvw[2 * i];
        float b = kv[2 * i + 1] * scale * kvw[2 * i + 1];
        kvnorm_bf[(size_t)row * (KV_LORA / 2) + i] = pack_bf16x2(a, b);
    }
    for (int idx = t; idx < NUM_HEADS * 32; idx += 256) {   // RoPE q_pe
        int h = idx >> 5, d = idx & 31;
        float c = cosp[row * 32 + d], sn = sinp[row * 32 + d];
        float* qp = q + (size_t)row * (NUM_HEADS * QK_HEAD) + h * QK_HEAD + QK_NOPE + 2 * d;
        float e = qp[0], o = qp[1];
        qp[0] = e * c - o * sn;
        qp[1] = e * sn + o * c;
    }
    if (t < 32) {                                           // RoPE k_pe
        int d = t;
        float c = cosp[row * 32 + d], sn = sinp[row * 32 + d];
        const float* kp = kv + KV_LORA + 2 * d;
        float e = kp[0], o = kp[1];
        kpe[(size_t)row * QK_ROPE + 2 * d]     = e * c - o * sn;
        kpe[(size_t)row * QK_ROPE + 2 * d + 1] = e * sn + o * c;
    }
}

// ---------------------------------------------------------------------------
// Pack attention operands to bf16, head-major:
//   qbf[h][s][128], kbf[h][s][128] (k_nope | broadcast k_pe), vbfT[h][d][s].
// V is pre-transposed so attention V-fragments are contiguous async loads.
// ---------------------------------------------------------------------------
__global__ __launch_bounds__(256)
void pack_qkv_kernel(const float* __restrict__ q,      // [SEQ, 2048] post-rope
                     const float* __restrict__ kvb,    // [SEQ, 3072]
                     const float* __restrict__ kpe,    // [SEQ, 64]
                     uint32_t* __restrict__ qbf,       // [16][SEQ][64] u32
                     uint32_t* __restrict__ kbf,       // [16][SEQ][64] u32
                     uint16_t* __restrict__ vbfT) {    // [16][128][SEQ] u16
    const int row = blockIdx.x;
    const int t   = threadIdx.x;
    for (int idx = t; idx < NUM_HEADS * 64; idx += 256) {
        int h = idx >> 6, c = idx & 63, d0 = 2 * c;
        const float* qs = q + (size_t)row * (NUM_HEADS * QK_HEAD) + h * QK_HEAD + d0;
        qbf[((size_t)h * SEQ + row) * 64 + c] = pack_bf16x2(qs[0], qs[1]);
        float ka, kb;
        if (d0 < QK_NOPE) {
            const float* s = kvb + (size_t)row * KVB_N + h * (QK_NOPE + V_HEAD) + d0;
            ka = s[0]; kb = s[1];
        } else {
            const float* s = kpe + (size_t)row * QK_ROPE + (d0 - QK_NOPE);
            ka = s[0]; kb = s[1];
        }
        kbf[((size_t)h * SEQ + row) * 64 + c] = pack_bf16x2(ka, kb);
    }
    for (int idx = t; idx < NUM_HEADS * V_HEAD; idx += 256) {
        int h = idx >> 7, d = idx & 127;
        float v = kvb[(size_t)row * KVB_N + h * (QK_NOPE + V_HEAD) + QK_NOPE + d];
        vbfT[((size_t)h * V_HEAD + d) * SEQ + row] = (uint16_t)f2bf(v);
    }
}

// ---------------------------------------------------------------------------
// Flash attention (causal). Block = (64 Q rows, head h), 4 waves, wave = 16 rows.
// bf16 K / V^T tiles double-buffered via async DMA; online softmax with 16-lane
// shuffles; P redistributed via per-wave LDS scratch (s_wait_dscnt 0).
// Output written as bf16 (A operand of the final GEMM).
// ---------------------------------------------------------------------------
__global__ __launch_bounds__(128)
void attn_kernel(const uint32_t* __restrict__ qbf,   // [16][SEQ][64]
                 const uint32_t* __restrict__ kbf,   // [16][SEQ][64]
                 const uint16_t* __restrict__ vbfT,  // [16][128][SEQ]
                 uint16_t* __restrict__ attn_bf) {   // [SEQ][2048]
    __shared__ uint32_t Qs[64 * 64];       // 64 x 128 bf16   (16 KB)
    __shared__ uint32_t Ks[2][32 * 64];    // 32 x 128 bf16   ( 8 KB x2)
    __shared__ uint32_t Vs[2][128 * 16];   // V^T 128 x 32    ( 8 KB x2)
    __shared__ uint32_t Ps[4][16 * 16];    // per-wave P      ( 4 KB)

    const int t    = threadIdx.x;
    const int w    = t >> 5;
    const int lane = t & 31;
    const int half = lane >> 4;
    const int n    = lane & 15;
    const int h    = blockIdx.y;
    const int qb   = blockIdx.x;
    const int r0   = qb * 64 + w * 16;

    const uint32_t* qh = qbf + (size_t)h * SEQ * 64;
    const uint32_t* kh = kbf + (size_t)h * SEQ * 64;
    const uint16_t* vh = vbfT + (size_t)h * V_HEAD * SEQ;

    auto load_K = [&](int kb, uint32_t* tile) {      // 32 rows x 256B
#pragma unroll
        for (int i = 0; i < 4; ++i) {
            int c = t + i * 128;
            int row = c >> 4, sub = c & 15;
            async_copy16(kh + (size_t)(kb + row) * 64 + sub * 4,
                         (char*)tile + row * 256 + sub * 16);
        }
    };
    auto load_V = [&](int kb, uint32_t* tile) {      // 128 rows x 64B of V^T
#pragma unroll
        for (int i = 0; i < 4; ++i) {
            int c = t + i * 128;
            int d = c >> 2, sub = c & 3;
            async_copy16(vh + (size_t)d * SEQ + kb + sub * 8,
                         (char*)tile + d * 64 + sub * 16);
        }
    };

    // stage Q tile (64 x 256B) + first K/V tiles
#pragma unroll
    for (int i = 0; i < 8; ++i) {
        int c = t + i * 128;
        int row = c >> 4, sub = c & 15;
        async_copy16(qh + (size_t)(qb * 64 + row) * 64 + sub * 4,
                     (char*)Qs + row * 256 + sub * 16);
    }
    load_K(0, Ks[0]);
    load_V(0, Vs[0]);
    wait_async0();
    __syncthreads();

    v16bf qf[4];
#pragma unroll
    for (int c = 0; c < 4; ++c)
        qf[c] = load_frag_lds(Qs + (w * 16) * 64 + c * 16, 64);

    v8f O[8];
#pragma unroll
    for (int d = 0; d < 8; ++d)
        O[d] = (v8f){0.f, 0.f, 0.f, 0.f, 0.f, 0.f, 0.f, 0.f};
    float m_run[8], l_run[8];
#pragma unroll
    for (int j = 0; j < 8; ++j) { m_run[j] = -3.0e38f; l_run[j] = 0.f; }

    const int kend = qb * 64 + 64;         // causal bound, uniform across block
    int buf = 0;
    for (int kb = 0; kb < kend; kb += 32) {
        if (kb + 32 < kend) {              // async-prefetch next K/V tiles
            load_K(kb + 32, Ks[buf ^ 1]);
            load_V(kb + 32, Vs[buf ^ 1]);
        }
        // S = Q @ K^T (two 16x16 tiles)
        v8f s0 = (v8f){0.f, 0.f, 0.f, 0.f, 0.f, 0.f, 0.f, 0.f};
        v8f s1 = (v8f){0.f, 0.f, 0.f, 0.f, 0.f, 0.f, 0.f, 0.f};
#pragma unroll
        for (int c = 0; c < 4; ++c) {
            s0 = wmma_bf16(qf[c], load_frag_lds(Ks[buf] + 0  * 64 + c * 16, 64), s0);
            s1 = wmma_bf16(qf[c], load_frag_lds(Ks[buf] + 16 * 64 + c * 16, 64), s1);
        }

        // scale + causal mask + online softmax
        float alphaj[8];
#pragma unroll
        for (int j = 0; j < 8; ++j) {
            int r = r0 + j + 8 * half;
            float v0 = s0[j] * ATT_SCALE + ((kb + n)      > r ? -1.0e9f : 0.f);
            float v1 = s1[j] * ATT_SCALE + ((kb + 16 + n) > r ? -1.0e9f : 0.f);
            float mx = fmaxf(v0, v1);
#pragma unroll
            for (int off = 1; off < 16; off <<= 1)
                mx = fmaxf(mx, __shfl_xor(mx, off, 32));
            float mnew  = fmaxf(m_run[j], mx);
            float alpha = __expf(m_run[j] - mnew);
            float p0 = __expf(v0 - mnew), p1 = __expf(v1 - mnew);
            float rs = p0 + p1;
#pragma unroll
            for (int off = 1; off < 16; off <<= 1)
                rs += __shfl_xor(rs, off, 32);
            l_run[j] = l_run[j] * alpha + rs;
            m_run[j] = mnew;
            alphaj[j] = alpha;
            s0[j] = p0; s1[j] = p1;
        }
#pragma unroll
        for (int d = 0; d < 8; ++d)
#pragma unroll
            for (int j = 0; j < 8; ++j)
                O[d][j] *= alphaj[j];

        // redistribute P (D layout) -> A-fragment layout through LDS
        unsigned short* ps = (unsigned short*)Ps[w];
#pragma unroll
        for (int j = 0; j < 8; ++j) {
            int r = j + 8 * half;
            ps[r * 32 + n]      = (unsigned short)f2bf(s0[j]);
            ps[r * 32 + 16 + n] = (unsigned short)f2bf(s1[j]);
        }
        asm volatile("s_wait_dscnt 0" ::: "memory");   // intra-wave DS RAW
        v16bf pf = load_frag_lds(Ps[w], 16);

        // O += P @ V  (V^T rows in Vs, stride 16 u32)
#pragma unroll
        for (int d = 0; d < 8; ++d)
            O[d] = wmma_bf16(pf, load_frag_lds(Vs[buf] + (d * 16) * 16, 16), O[d]);

        wait_async0();
        __syncthreads();
        buf ^= 1;
    }

    // normalize + store bf16
#pragma unroll
    for (int d = 0; d < 8; ++d)
#pragma unroll
        for (int j = 0; j < 8; ++j) {
            int r = r0 + j + 8 * half;
            attn_bf[(size_t)r * (NUM_HEADS * V_HEAD) + h * V_HEAD + d * 16 + n] =
                (uint16_t)f2bf(O[d][j] / l_run[j]);
        }
}

// ---------------------------------------------------------------------------
extern "C" void kernel_launch(void* const* d_in, const int* in_sizes, int n_in,
                              void* d_out, int out_size, void* d_ws, size_t ws_size,
                              hipStream_t stream) {
    const float* x     = (const float*)d_in[0];
    const float* cosp  = (const float*)d_in[1];
    const float* sinp  = (const float*)d_in[2];
    // d_in[3] = mask (causal applied analytically), d_in[9] = start_pos (0)
    const float* wq    = (const float*)d_in[4];
    const float* wkv_a = (const float*)d_in[5];
    const float* kvw   = (const float*)d_in[6];
    const float* wkv_b = (const float*)d_in[7];
    const float* wo    = (const float*)d_in[8];
    float* out = (float*)d_out;

    char* p = (char*)d_ws;
    auto take = [&](size_t bytes) { char* r = p; p += bytes; return r; };
    uint32_t* xbf     = (uint32_t*)take((size_t)SEQ * D_IN * 2);          // bf16
    uint32_t* wqbf    = (uint32_t*)take((size_t)D_IN * D_IN * 2);
    uint32_t* wkvabf  = (uint32_t*)take((size_t)KVA_N * D_IN * 2);
    uint32_t* wkvbbf  = (uint32_t*)take((size_t)KVB_N * KV_LORA * 2);
    uint32_t* wobf    = (uint32_t*)take((size_t)D_IN * D_IN * 2);
    float*    qbuf    = (float*)take((size_t)SEQ * D_IN * 4);
    float*    kvfull  = (float*)take((size_t)SEQ * KVA_N * 4);
    uint32_t* kvnormb = (uint32_t*)take((size_t)SEQ * KV_LORA * 2);
    float*    kpe     = (float*)take((size_t)SEQ * QK_ROPE * 4);
    float*    kvb     = (float*)take((size_t)SEQ * KVB_N * 4);
    uint32_t* qbf     = (uint32_t*)take((size_t)NUM_HEADS * SEQ * QK_HEAD * 2);
    uint32_t* kbf     = (uint32_t*)take((size_t)NUM_HEADS * SEQ * QK_HEAD * 2);
    uint16_t* vbfT    = (uint16_t*)take((size_t)NUM_HEADS * V_HEAD * SEQ * 2);
    uint16_t* attnbf  = (uint16_t*)take((size_t)SEQ * D_IN * 2);

    auto cvt = [&](const float* s, uint32_t* d, size_t elems) {
        int np = (int)(elems / 2);
        cvt_f32_bf16<<<(np + 255) / 256, 256, 0, stream>>>(s, d, np);
    };
    cvt(x,     xbf,    (size_t)SEQ * D_IN);
    cvt(wq,    wqbf,   (size_t)D_IN * D_IN);
    cvt(wkv_a, wkvabf, (size_t)KVA_N * D_IN);
    cvt(wkv_b, wkvbbf, (size_t)KVB_N * KV_LORA);
    cvt(wo,    wobf,   (size_t)D_IN * D_IN);

    dim3 blk(128);
    gemm_bf16_wmma<<<dim3(D_IN / 64, SEQ / 64), blk, 0, stream>>>(
        (const uint16_t*)xbf, (const uint16_t*)wqbf, qbuf, SEQ, D_IN, D_IN);
    gemm_bf16_wmma<<<dim3(KVA_N / 64, SEQ / 64), blk, 0, stream>>>(
        (const uint16_t*)xbf, (const uint16_t*)wkvabf, kvfull, SEQ, KVA_N, D_IN);
    rms_rope_kernel<<<SEQ, 256, 0, stream>>>(qbuf, kvfull, cosp, sinp, kvw, kvnormb, kpe);
    gemm_bf16_wmma<<<dim3(KVB_N / 64, SEQ / 64), blk, 0, stream>>>(
        (const uint16_t*)kvnormb, (const uint16_t*)wkvbbf, kvb, SEQ, KVB_N, KV_LORA);
    pack_qkv_kernel<<<SEQ, 256, 0, stream>>>(qbuf, kvb, kpe, qbf, kbf, vbfT);
    attn_kernel<<<dim3(SEQ / 64, NUM_HEADS), blk, 0, stream>>>(qbf, kbf, vbfT, attnbf);
    gemm_bf16_wmma<<<dim3(D_IN / 64, SEQ / 64), blk, 0, stream>>>(
        attnbf, (const uint16_t*)wobf, out, SEQ, D_IN, NUM_HEADS * V_HEAD);
}
